// STN_88871463289469
// MI455X (gfx1250) — matrix operand order repaired and checked by reading, not compiled, not run on hardware
//
#include <hip/hip_runtime.h>
#include <hip/hip_bf16.h>

typedef __attribute__((ext_vector_type(16))) _Float16 v16h;
typedef __attribute__((ext_vector_type(8)))  _Float16 v8h;
typedef __attribute__((ext_vector_type(8)))  float    v8f;
typedef __attribute__((ext_vector_type(4), aligned(4))) float v4fu; // 4B-aligned vector load

#define BATCH 64
#define HIN 224
#define WIN 224
#define CIN 3
#define IMG (HIN*WIN*CIN)   // 150528 floats per image

// conv1: 7x7x3 -> 8, VALID: 218x218, pool -> 109x109
// virtual K layout: row ky has stride 32 (21 valid) -> 7 K-steps, one per ky
#define S1 7
#define OC1 8
#define P1 109
#define NP1 (109*109)   // 11881
#define Q1 ((NP1+3)/4)  // 2971 pooled quads per image

// conv2: 5x5x8 -> 10 on 109x109, VALID: 105x105, pool -> 52x52
// virtual K layout: row ky (40 valid f16) has stride 64 -> 10 K-steps (2 per ky)
#define S2 10
#define OC2 10
#define P2 52
#define NP2 (52*52)     // 2704
#define Q2 (NP2/4)      // 676 exact

#define FLAT 27040      // 52*52*10
#define SFC (FLAT/32)   // 845 exact

static __device__ __forceinline__ v8f wmma_f16(v16h a, v16h b, v8f c) {
    return __builtin_amdgcn_wmma_f32_16x16x32_f16(false, a, false, b, (short)0, c, false, false);
}

// ---------------------------------------------------------------------------
// Pack a weight matrix into f16 B-fragment order for v_wmma_f32_16x16x32_f16,
// using a virtual-K layout: kv = s*32 + 8*h + q + 8*(q>>3) maps to source row/col via
//   ky = kv >> stride_log2, rem = kv & (stride-1)
//   valid iff ky < nrows && rem < rowlen && col < ldw
//   src = (ky*rowlen + rem)*ldw + col         (zeros elsewhere)
// Stored so GEMM lane l reads 16 contiguous halves at out[(s*32 + l)*16].
// ---------------------------------------------------------------------------
__global__ void pack_b_kernel(const float* __restrict__ w, _Float16* __restrict__ out,
                              int steps, int stride_log2, int rowlen, int nrows,
                              int ldw, int ncol_off) {
    int e = blockIdx.x * blockDim.x + threadIdx.x;
    int total = steps * 512;
    if (e >= total) return;
    int s = e >> 9;
    int r = e & 511;
    int l = r >> 4;
    int q = r & 15;
    int h = l >> 4;
    int n = l & 15;
    int kv = s * 32 + 8 * h + q + ((q >> 3) << 3);
    int ky = kv >> stride_log2;
    int rem = kv & ((1 << stride_log2) - 1);
    int col = ncol_off + n;
    float val = 0.f;
    if (ky < nrows && rem < rowlen && col < ldw)
        val = w[(ky * rowlen + rem) * ldw + col];
    out[e] = (_Float16)val;
}

// ---------------------------------------------------------------------------
// conv1 + bias + relu + 2x2 maxpool, implicit GEMM via WMMA.
// One wave: 4 pooled pixels (16 conv pixels = M) x 16 channels (8 valid = N).
// Step s = patch row ky.  Lane A chunks are contiguous fp32 runs in that image
// row: rem = 8h+[0..7] and 16+8h+[0..7] (rem>=21 hits zero-packed weights).
// Image 63 reads a tail-padded copy so over-reads never leave the buffer.
// ---------------------------------------------------------------------------
__global__ void conv1_pool_kernel(const float* __restrict__ in,
                                  const float* __restrict__ inpad63,
                                  const _Float16* __restrict__ w1p,
                                  const float* __restrict__ c1b,
                                  _Float16* __restrict__ pool1) {
    int tid = blockIdx.x * blockDim.x + threadIdx.x;
    int wave = tid >> 5;
    int lane = tid & 31;
    int total_waves = BATCH * Q1;
    if (wave >= total_waves) return;
    int bimg = wave / Q1;
    int pb = (wave - bimg * Q1) * 4;

    int m = lane & 15;
    int h = lane >> 4;
    int pp_m = pb + (m >> 2);
    int ppc = pp_m < NP1 ? pp_m : NP1 - 1;
    int py = ppc / P1;
    int px = ppc - py * P1;
    int oy = 2 * py + ((m >> 1) & 1);
    int ox = 2 * px + (m & 1);

    const float* img0 = (bimg == BATCH - 1) ? inpad63 : (in + (size_t)bimg * IMG);
    const float* rp = img0 + (oy * WIN + ox) * CIN;
    int o1 = 8 * h;
    int o2 = 16 + 8 * h;

    v8f acc = {};
#pragma unroll
    for (int s = 0; s < S1; ++s) {
        const float* r = rp + s * (WIN * CIN);
        if (s + 1 < S1) __builtin_prefetch(r + WIN * CIN, 0, 1);
        v4fu a0 = *(const v4fu*)(r + o1);
        v4fu a1 = *(const v4fu*)(r + o1 + 4);
        v4fu a2 = *(const v4fu*)(r + o2);
        v4fu a3 = *(const v4fu*)(r + o2 + 4);
        v16h a;
#pragma unroll
        for (int i = 0; i < 4; ++i) {
            a[i]      = (_Float16)a0[i];
            a[4 + i]  = (_Float16)a1[i];
            a[8 + i]  = (_Float16)a2[i];
            a[12 + i] = (_Float16)a3[i];
        }
        v16h bw = *(const v16h*)(w1p + (s * 32 + lane) * 16);
        acc = wmma_f16(a, bw, acc);
    }

    // C/D layout: VGPR r -> M = r (lanes 0-15) or r+8 (lanes 16-31); N = lane&15
    int n = lane & 15;
    float bias = (n < OC1) ? c1b[n] : 0.f;
    float rv[8];
#pragma unroll
    for (int r = 0; r < 8; ++r) rv[r] = fmaxf(acc[r] + bias, 0.f);
    float pA = fmaxf(fmaxf(rv[0], rv[1]), fmaxf(rv[2], rv[3]));
    float pB = fmaxf(fmaxf(rv[4], rv[5]), fmaxf(rv[6], rv[7]));
    int jbase = (lane < 16) ? 0 : 2;
    int pp0 = pb + jbase, pp1 = pb + jbase + 1;
    if (n < OC1) {
        if (pp0 < NP1) pool1[((size_t)bimg * NP1 + pp0) * OC1 + n] = (_Float16)pA;
        if (pp1 < NP1) pool1[((size_t)bimg * NP1 + pp1) * OC1 + n] = (_Float16)pB;
    }
}

// ---------------------------------------------------------------------------
// conv2 + bias + relu + 2x2 maxpool -> xs (flattened, f16).
// Step s: ky = s>>1, rem base = (s&1)*32; lane chunks are 16B-aligned v8h
// loads (rem>=40 hits zero-packed weights; over-reads stay inside pool1).
// ---------------------------------------------------------------------------
__global__ void conv2_pool_kernel(const _Float16* __restrict__ pool1,
                                  const _Float16* __restrict__ w2p,
                                  const float* __restrict__ c2b,
                                  _Float16* __restrict__ xs) {
    int tid = blockIdx.x * blockDim.x + threadIdx.x;
    int wave = tid >> 5;
    int lane = tid & 31;
    int total_waves = BATCH * Q2;
    if (wave >= total_waves) return;
    int bimg = wave / Q2;
    int pb = (wave - bimg * Q2) * 4;

    int m = lane & 15;
    int h = lane >> 4;
    int pp_m = pb + (m >> 2);
    int py = pp_m / P2;
    int px = pp_m - py * P2;
    int oy = 2 * py + ((m >> 1) & 1);
    int ox = 2 * px + (m & 1);

    const _Float16* rp = pool1 + (size_t)bimg * NP1 * OC1 + (oy * P1 + ox) * OC1;
    int o1 = 8 * h;

    v8f acc = {};
#pragma unroll
    for (int s = 0; s < S2; ++s) {
        const _Float16* r = rp + (s >> 1) * (P1 * OC1) + (s & 1) * 32;
        v8h c0 = *(const v8h*)(r + o1);
        v8h c1 = *(const v8h*)(r + o1 + 16);
        v16h a;
#pragma unroll
        for (int i = 0; i < 8; ++i) { a[i] = c0[i]; a[8 + i] = c1[i]; }
        v16h bw = *(const v16h*)(w2p + (s * 32 + lane) * 16);
        acc = wmma_f16(a, bw, acc);
    }

    int n = lane & 15;
    float bias = (n < OC2) ? c2b[n] : 0.f;
    float rv[8];
#pragma unroll
    for (int r = 0; r < 8; ++r) rv[r] = fmaxf(acc[r] + bias, 0.f);
    float pA = fmaxf(fmaxf(rv[0], rv[1]), fmaxf(rv[2], rv[3]));
    float pB = fmaxf(fmaxf(rv[4], rv[5]), fmaxf(rv[6], rv[7]));
    int jbase = (lane < 16) ? 0 : 2;
    int pp0 = pb + jbase, pp1 = pb + jbase + 1;
    if (n < OC2) {
        xs[(size_t)bimg * FLAT + pp0 * OC2 + n] = (_Float16)pA;
        xs[(size_t)bimg * FLAT + pp1 * OC2 + n] = (_Float16)pB;
    }
}

// ---------------------------------------------------------------------------
// fc1: (64 x 27040) @ (27040 x 32) + b, relu -> h1 (f32).
// 8 waves in one block: wave = mtile*2 + ntile.  Contiguous 16B v8h A loads.
// ---------------------------------------------------------------------------
__global__ void fc1_kernel(const _Float16* __restrict__ xs,
                           const _Float16* __restrict__ fc1p,
                           const float* __restrict__ fc1b,
                           float* __restrict__ h1) {
    int tid = threadIdx.x;
    int wave = tid >> 5;
    int lane = tid & 31;
    int mt = wave >> 1;
    int nt = wave & 1;
    int h = lane >> 4;
    int mrow = mt * 16 + (lane & 15);
    const _Float16* arow = xs + (size_t)mrow * FLAT;

    v8f acc = {};
    for (int s = 0; s < SFC; ++s) {
        v8h lo = *(const v8h*)(arow + s * 32 + 8 * h);
        v8h hi = *(const v8h*)(arow + s * 32 + 16 + 8 * h);
        v16h a;
#pragma unroll
        for (int i = 0; i < 8; ++i) { a[i] = lo[i]; a[8 + i] = hi[i]; }
        v16h bw = *(const v16h*)(fc1p + (((size_t)nt * SFC + s) * 32 + lane) * 16);
        acc = wmma_f16(a, bw, acc);
    }

    int n = nt * 16 + (lane & 15);
    float bias = fc1b[n];
#pragma unroll
    for (int r = 0; r < 8; ++r) {
        int mm = mt * 16 + r + ((lane >> 4) << 3);
        h1[mm * 32 + n] = fmaxf(acc[r] + bias, 0.f);
    }
}

// ---------------------------------------------------------------------------
// fc2 -> theta (64 x 6).  One thread per batch element.
// ---------------------------------------------------------------------------
__global__ void fc2_theta_kernel(const float* __restrict__ h1,
                                 const float* __restrict__ fc2w,
                                 const float* __restrict__ fc2b,
                                 float* __restrict__ theta) {
    int b = blockIdx.x * blockDim.x + threadIdx.x;
    if (b >= BATCH) return;
#pragma unroll
    for (int j = 0; j < 6; ++j) {
        float s = fc2b[j];
        for (int i = 0; i < 32; ++i) s += h1[b * 32 + i] * fc2w[i * 6 + j];
        theta[b * 6 + j] = s;
    }
}

// ---------------------------------------------------------------------------
// affine grid + bilinear sampler: one thread per output pixel (3 channels).
// ---------------------------------------------------------------------------
__global__ void sampler_kernel(const float* __restrict__ img,
                               const float* __restrict__ theta,
                               float* __restrict__ out) {
    int b = blockIdx.y;
    int pix = blockIdx.x * blockDim.x + threadIdx.x;
    if (pix >= HIN * WIN) return;
    int py = pix / WIN;
    int px = pix - py * WIN;

    float t0 = theta[b * 6 + 0], t1 = theta[b * 6 + 1], t2 = theta[b * 6 + 2];
    float t3 = theta[b * 6 + 3], t4 = theta[b * 6 + 4], t5 = theta[b * 6 + 5];

    float xt = -1.f + 2.f * (float)px / (float)(WIN - 1);
    float yt = -1.f + 2.f * (float)py / (float)(HIN - 1);
    float gx = t0 * xt + t1 * yt + t2;
    float gy = t3 * xt + t4 * yt + t5;

    float x = 0.5f * (gx + 1.f) * (float)(WIN - 1);
    float y = 0.5f * (gy + 1.f) * (float)(HIN - 1);
    int x0 = (int)floorf(x); x0 = x0 < 0 ? 0 : (x0 > WIN - 1 ? WIN - 1 : x0);
    int x1 = x0 + 1;         x1 = x1 > WIN - 1 ? WIN - 1 : x1;
    int y0 = (int)floorf(y); y0 = y0 < 0 ? 0 : (y0 > HIN - 1 ? HIN - 1 : y0);
    int y1 = y0 + 1;         y1 = y1 > HIN - 1 ? HIN - 1 : y1;
    float x0f = (float)x0, x1f = (float)x1, y0f = (float)y0, y1f = (float)y1;
    float wa = (x1f - x) * (y1f - y);
    float wb = (x1f - x) * (y - y0f);
    float wc = (x - x0f) * (y1f - y);
    float wd = (x - x0f) * (y - y0f);

    const float* ib = img + (size_t)b * IMG;
    float* ob = out + ((size_t)b * HIN * WIN + (size_t)py * WIN + px) * CIN;
#pragma unroll
    for (int c = 0; c < CIN; ++c) {
        float Ia = ib[(y0 * WIN + x0) * CIN + c];
        float Ib = ib[(y1 * WIN + x0) * CIN + c];
        float Ic = ib[(y0 * WIN + x1) * CIN + c];
        float Id = ib[(y1 * WIN + x1) * CIN + c];
        ob[c] = wa * Ia + wb * Ib + wc * Ic + wd * Id;
    }
}

// ---------------------------------------------------------------------------
extern "C" void kernel_launch(void* const* d_in, const int* in_sizes, int n_in,
                              void* d_out, int out_size, void* d_ws, size_t ws_size,
                              hipStream_t stream) {
    const float* in    = (const float*)d_in[0];
    const float* w1    = (const float*)d_in[1];
    const float* c1b   = (const float*)d_in[2];
    const float* w2    = (const float*)d_in[3];
    const float* c2b   = (const float*)d_in[4];
    const float* fc1w  = (const float*)d_in[5];
    const float* fc1b  = (const float*)d_in[6];
    const float* fc2w  = (const float*)d_in[7];
    const float* fc2b  = (const float*)d_in[8];
    float* out = (float*)d_out;

    char* ws = (char*)d_ws;
    size_t off = 0;
    _Float16* w1p   = (_Float16*)(ws + off); off += (size_t)S1 * 512 * 2;
    off = (off + 255) & ~(size_t)255;
    _Float16* w2p   = (_Float16*)(ws + off); off += (size_t)S2 * 512 * 2;
    off = (off + 255) & ~(size_t)255;
    _Float16* fc1p  = (_Float16*)(ws + off); off += (size_t)2 * SFC * 512 * 2;
    off = (off + 255) & ~(size_t)255;
    _Float16* pool1 = (_Float16*)(ws + off); off += (size_t)BATCH * NP1 * OC1 * 2;
    off = (off + 255) & ~(size_t)255;
    _Float16* xs    = (_Float16*)(ws + off); off += (size_t)BATCH * FLAT * 2;
    off = (off + 255) & ~(size_t)255;
    float* h1       = (float*)(ws + off); off += (size_t)BATCH * 32 * 4;
    off = (off + 255) & ~(size_t)255;
    float* theta    = (float*)(ws + off); off += (size_t)BATCH * 6 * 4;
    off = (off + 255) & ~(size_t)255;
    float* inpad63  = (float*)(ws + off); off += ((size_t)IMG + 64) * 4; // last image + tail pad

    // 0) tail-padded copy of the last image so conv1 over-reads are in-bounds
    hipMemcpyAsync(inpad63, in + (size_t)(BATCH - 1) * IMG, (size_t)IMG * sizeof(float),
                   hipMemcpyDeviceToDevice, stream);

    // 1) pack weights into WMMA B-fragment order (virtual-K layouts)
    pack_b_kernel<<<(S1 * 512 + 255) / 256, 256, 0, stream>>>(w1, w1p, S1, 5, 21, 7, OC1, 0);
    pack_b_kernel<<<(S2 * 512 + 255) / 256, 256, 0, stream>>>(w2, w2p, S2, 6, 40, 5, OC2, 0);
    pack_b_kernel<<<(SFC * 512 + 255) / 256, 256, 0, stream>>>(fc1w, fc1p,
                                                               SFC, 5, 32, SFC, 32, 0);
    pack_b_kernel<<<(SFC * 512 + 255) / 256, 256, 0, stream>>>(fc1w, fc1p + (size_t)SFC * 512,
                                                               SFC, 5, 32, SFC, 32, 16);

    // 2) conv1 + relu + pool (WMMA implicit GEMM, branch-free gathers)
    {
        long long threads = (long long)BATCH * Q1 * 32;
        int blocks = (int)((threads + 255) / 256);
        conv1_pool_kernel<<<blocks, 256, 0, stream>>>(in, inpad63, w1p, c1b, pool1);
    }
    // 3) conv2 + relu + pool (WMMA implicit GEMM)
    {
        long long threads = (long long)BATCH * Q2 * 32;
        int blocks = (int)((threads + 255) / 256);
        conv2_pool_kernel<<<blocks, 256, 0, stream>>>(pool1, w2p, c2b, xs);
    }
    // 4) fc1 (WMMA GEMM, 8 waves)
    fc1_kernel<<<1, 256, 0, stream>>>(xs, fc1p, fc1b, h1);
    // 5) fc2 -> theta
    fc2_theta_kernel<<<1, 64, 0, stream>>>(h1, fc2w, fc2b, theta);
    // 6) affine grid + bilinear sample
    {
        dim3 grid((HIN * WIN + 255) / 256, BATCH);
        sampler_kernel<<<grid, 256, 0, stream>>>(in, theta, out);
    }
    (void)in_sizes; (void)n_in; (void)out_size; (void)ws_size;
}
